// DecoderLSTM_1408749273722
// MI455X (gfx1250) — matrix operand order repaired and checked by reading, not compiled, not run on hardware
//
#include <hip/hip_runtime.h>
#include <hip/hip_bf16.h>

// ---------------------------------------------------------------------------
// DecoderLSTM on gfx1250 (MI455X), bf16 WMMA pipeline with async-LDS staging.
// B=64, U=1024, E=V=8000, T=64.
// ---------------------------------------------------------------------------

typedef __attribute__((ext_vector_type(16))) __bf16 bf16x16;
typedef __attribute__((ext_vector_type(8)))  __bf16 bf16x8;
typedef __attribute__((ext_vector_type(8)))  float  f32x8;
typedef __attribute__((ext_vector_type(4)))  int    v4i;

#define LSTM_B 64
#define LSTM_U 1024
#define LSTM_V 8000
#define LSTM_E 8000
#define LSTM_T 64
#define GATES_N (4*LSTM_U)          // 4096

// LDS A-slice: 64 rows x 32 bf16, row stride padded to 80B -> conflict-free
// half-wave ds_load_b128 (16 lanes x 4 words cover all 64 banks exactly).
#define A_ROW_BYTES 80
#define A_BUF_BYTES (64 * A_ROW_BYTES)     // one 64x32 k-tile slice (5120 B)
#define ST_BYTES    (2 * A_BUF_BYTES)      // super-tile = 2 k-tiles (10240 B)

#if defined(__has_builtin)
#if __has_builtin(__builtin_amdgcn_global_load_async_to_lds_b128) && \
    __has_builtin(__builtin_amdgcn_s_wait_asynccnt)
#define USE_ASYNC_LDS 1
#endif
#endif
#ifndef USE_ASYNC_LDS
#define USE_ASYNC_LDS 0
#endif

// ---------------------------------------------------------------------------
// Pack fp32 row-major (K x Nloc) weights into bf16 WMMA B-tiles.
// Tile (nTileGlobal, kt) at ((nTileGlobal*ktiles)+kt)*512 elems; inside a tile
// lane L holds 16 contiguous bf16: column n = nTile*16 + (L%16),
// k = kt*32 + (L/16)*16 + e  (this code's contiguous k order: lanes 0-15 K=0..15, 16-31 K=16..31).
// ---------------------------------------------------------------------------
__global__ void pack_weights(const float* __restrict__ W, __bf16* __restrict__ P,
                             int K, int Nloc, int ntOffset, int ktiles,
                             long long elems) {
  long long idx = (long long)blockIdx.x * blockDim.x + threadIdx.x;
  if (idx >= elems) return;
  int n = (int)(idx % Nloc);
  int k = (int)(idx / Nloc);
  int nt = n >> 4, nl = n & 15;
  int kt = k >> 5, ko = k & 31;
  int hi = ko >> 4;
  int e  = ko & 15;
  int lane = hi * 16 + nl;
  size_t dst = ((size_t)(ntOffset + nt) * ktiles + kt) * 512 + lane * 16 + e;
  P[dst] = (__bf16)W[idx];
}

__global__ void concat_bias(const float* __restrict__ b0, const float* __restrict__ b1,
                            const float* __restrict__ b2, const float* __restrict__ b3,
                            float* __restrict__ bias4) {
  int idx = blockIdx.x * blockDim.x + threadIdx.x;
  if (idx >= GATES_N) return;
  int g = idx >> 10, u = idx & 1023;
  const float* src = (g == 0) ? b0 : (g == 1) ? b1 : (g == 2) ? b2 : b3;
  bias4[idx] = src[u];
}

__global__ void init_state(const float* __restrict__ y, __bf16* __restrict__ outB,
                           float* __restrict__ c, __bf16* __restrict__ hB) {
  int idx = blockIdx.x * blockDim.x + threadIdx.x;
  if (idx < LSTM_B * LSTM_E) outB[idx] = (__bf16)y[idx];
  if (idx < LSTM_B * LSTM_U) { c[idx] = 0.0f; hB[idx] = (__bf16)0.0f; }
}

// Stage one 16-byte chunk of a 64x32 A slice into LDS (per thread).
__device__ __forceinline__ void stage_chunk(const __bf16* __restrict__ gsrc,
                                            unsigned char* ldst) {
#if USE_ASYNC_LDS
  // (int4 addrspace(1)*, int4 addrspace(3)*, imm offset, imm cpol)
  // generic->LDS offset is the low 32 bits of the generic address.
  typedef __attribute__((address_space(1))) v4i* gp_v4i;
  typedef __attribute__((address_space(3))) v4i* lp_v4i;
  __builtin_amdgcn_global_load_async_to_lds_b128(
      (gp_v4i)(uintptr_t)gsrc, (lp_v4i)(uintptr_t)ldst, 0, 0);
#else
  *(bf16x8*)ldst = *(const bf16x8*)gsrc;
#endif
}

// Wait until at most the next super-tile's 2 async loads remain in flight.
__device__ __forceinline__ void wait_stage_partial() {
#if USE_ASYNC_LDS
  __builtin_amdgcn_s_wait_asynccnt(2);
#endif
}

// ---------------------------------------------------------------------------
// WMMA GEMM: C(64 x N) = act( A1 @ B1 + A2 @ B2 + bias )
// A* bf16 row-major (lda = 32*kt*), B* packed bf16 tiles. 8 waves/block, each
// wave owns one 16-col N-tile and all four 16-row M-tiles.
// Pipeline: super-tiles of 2 k-tiles, triple-buffered async global->LDS A
// staging (partial s_wait_asynccnt keeps one super-tile in flight), B weight
// tiles prefetched one super-tile ahead in registers, A fragments preloaded
// into disjoint registers so the 4 WMMAs issue back-to-back into the XDL pipe.
// act: 0 = none, 1 = sigmoid.
// ---------------------------------------------------------------------------
__global__ void __launch_bounds__(256)
wmma_gemm_bias_act(const __bf16* __restrict__ A1, const __bf16* __restrict__ B1, int kt1,
                   const __bf16* __restrict__ A2, const __bf16* __restrict__ B2, int kt2,
                   const float* __restrict__ bias, float* __restrict__ C,
                   int N, int numNTiles, int act) {
  __shared__ __align__(16) unsigned char smemA[3][ST_BYTES];

  const int tid    = threadIdx.x;
  const int wave   = tid >> 5;
  const int lane   = tid & 31;
  const int laneHi = lane >> 4;
  const int laneLo = lane & 15;
  const int nTile  = blockIdx.x * 8 + wave;
  const bool active = (nTile < numNTiles);   // wave-uniform

  const int srow = tid >> 2;     // staging: 4 threads x 16B per 64B row
  const int sq   = tid & 3;
  const size_t soff = (size_t)(srow * A_ROW_BYTES + sq * 16);

  f32x8 acc[4] = {};

  // One 16x16x32 WMMA k-tile against all four M-tiles. A fragments are
  // preloaded into distinct registers first so the 8 ds_load_b128 issue as a
  // clause (single dscnt wait) and the 4 WMMAs chain only on the accumulators.
  auto compute_ktile = [&](const unsigned char* sbuf, bf16x16 bmat) {
    const unsigned char* abase = sbuf + laneHi * 32;
    union { bf16x16 v; bf16x8 h[2]; } a[4];
    #pragma unroll
    for (int m = 0; m < 4; ++m) {
      // A 16x32 bf16 fragment: lane row = lane%16; elems 0..15 = K b..b+15
      // with b = (lane/16)*16.
      const unsigned char* ar = abase + (size_t)(m * 16 + laneLo) * A_ROW_BYTES;
      a[m].h[0] = *(const bf16x8*)(ar);
      a[m].h[1] = *(const bf16x8*)(ar + 16);
    }
    #pragma unroll
    for (int m = 0; m < 4; ++m)
      acc[m] = __builtin_amdgcn_wmma_f32_16x16x32_bf16(
          false, a[m].v, false, bmat, (short)0, acc[m], false, false);
  };

  auto run_phase = [&](const __bf16* __restrict__ A, const __bf16* __restrict__ Bp,
                       int ktiles) {
    const int nst = ktiles >> 1;           // super-tiles (ktiles is even)
    const int lda = ktiles * 32;
    const __bf16* grow = A + (size_t)srow * lda + sq * 8;
    const __bf16* bptr = active ? (Bp + (size_t)nTile * ktiles * 512 + lane * 16)
                                : nullptr;

    // stage super-tile st into LDS buffer bufIdx (2 async b128 per thread)
    auto stage_st = [&](int st, int bufIdx) {
      unsigned char* base = smemA[bufIdx];
      const __bf16* g = grow + (size_t)st * 64;
      stage_chunk(g,      base + soff);
      stage_chunk(g + 32, base + A_BUF_BYTES + soff);
    };

    __syncthreads();                       // prior readers of all bufs done
    stage_st(0, 0);
    if (nst > 1) stage_st(1, 1);

    bf16x16 bcur0, bcur1;
    if (active) {
      bcur0 = *(const bf16x16*)(bptr);
      bcur1 = *(const bf16x16*)(bptr + 512);
    }

    int buf = 0;
    for (int st = 0; st < nst; ++st) {
      wait_stage_partial();                // current super-tile landed
      __syncthreads();                     // all waves staged + prev reads done
      if (st + 2 < nst) {
        int nb = buf + 2; if (nb >= 3) nb -= 3;
        stage_st(st + 2, nb);
      }
      if (active) {
        bf16x16 bn0 = bcur0, bn1 = bcur1;
        if (st + 1 < nst) {                // prefetch next super-tile's weights
          bn0 = *(const bf16x16*)(bptr + (size_t)(2 * st + 2) * 512);
          bn1 = *(const bf16x16*)(bptr + (size_t)(2 * st + 3) * 512);
        }
        compute_ktile(smemA[buf],               bcur0);
        compute_ktile(smemA[buf] + A_BUF_BYTES, bcur1);
        bcur0 = bn0; bcur1 = bn1;
      }
      if (++buf >= 3) buf = 0;
    }
  };

  run_phase(A1, B1, kt1);
  if (A2 != nullptr) run_phase(A2, B2, kt2);

  if (active) {
    const int col = nTile * 16 + laneLo;
    const float bv = bias ? bias[col] : 0.0f;
    #pragma unroll
    for (int m = 0; m < 4; ++m) {
      #pragma unroll
      for (int r = 0; r < 8; ++r) {
        // C 16x16 f32: VGPR r -> M = r (lanes 0-15) / r+8 (lanes 16-31)
        int row = m * 16 + laneHi * 8 + r;
        float v = acc[m][r] + bv;
        if (act == 1) v = 1.0f / (1.0f + __expf(-v));
        C[(size_t)row * N + col] = v;
      }
    }
  }
}

// ---------------------------------------------------------------------------
// c = f*c + i*g ; h = tanh(c)*o  (gate column order: f, i, o, g)
// ---------------------------------------------------------------------------
__global__ void cell_update(const float* __restrict__ gates, float* __restrict__ c,
                            __bf16* __restrict__ hB) {
  int idx = blockIdx.x * blockDim.x + threadIdx.x;
  if (idx >= LSTM_B * LSTM_U) return;
  int b = idx >> 10, u = idx & 1023;
  const float* g = gates + (size_t)b * GATES_N;
  float f  = g[u];
  float i  = g[LSTM_U + u];
  float o  = g[2 * LSTM_U + u];
  float gg = g[3 * LSTM_U + u];
  float cn = f * c[idx] + i * gg;
  c[idx] = cn;
  hB[idx] = (__bf16)(tanhf(cn) * o);
}

// ---------------------------------------------------------------------------
// Row softmax over V=8000; writes fp32 output slice [b, t, :] and bf16 feedback.
// ---------------------------------------------------------------------------
__global__ void __launch_bounds__(256)
softmax_row(const float* __restrict__ logits, float* __restrict__ outF,
            __bf16* __restrict__ outB, int t) {
  __shared__ float red[256];
  const int b = blockIdx.x;
  const float* row = logits + (size_t)b * LSTM_V;

  float m = -3.402823466e38f;
  for (int v = threadIdx.x; v < LSTM_V; v += 256) m = fmaxf(m, row[v]);
  red[threadIdx.x] = m;
  __syncthreads();
  for (int s = 128; s > 0; s >>= 1) {
    if (threadIdx.x < s) red[threadIdx.x] = fmaxf(red[threadIdx.x], red[threadIdx.x + s]);
    __syncthreads();
  }
  m = red[0];
  __syncthreads();

  float sum = 0.0f;
  for (int v = threadIdx.x; v < LSTM_V; v += 256) sum += __expf(row[v] - m);
  red[threadIdx.x] = sum;
  __syncthreads();
  for (int s = 128; s > 0; s >>= 1) {
    if (threadIdx.x < s) red[threadIdx.x] += red[threadIdx.x + s];
    __syncthreads();
  }
  const float inv = 1.0f / red[0];

  float* dst  = outF + ((size_t)b * LSTM_T + t) * LSTM_V;
  __bf16* dB  = outB + (size_t)b * LSTM_V;
  for (int v = threadIdx.x; v < LSTM_V; v += 256) {
    float p = __expf(row[v] - m) * inv;
    dst[v] = p;
    dB[v]  = (__bf16)p;
  }
}

// ---------------------------------------------------------------------------
// Host orchestration (graph-capture safe: only kernel launches on stream).
// ---------------------------------------------------------------------------
extern "C" void kernel_launch(void* const* d_in, const int* in_sizes, int n_in,
                              void* d_out, int out_size, void* d_ws, size_t ws_size,
                              hipStream_t stream) {
  (void)in_sizes; (void)n_in; (void)out_size; (void)ws_size;

  const float* y  = (const float*)d_in[0];
  const float* Wx[4] = {(const float*)d_in[1], (const float*)d_in[4],
                        (const float*)d_in[7], (const float*)d_in[10]};
  const float* Wh[4] = {(const float*)d_in[2], (const float*)d_in[5],
                        (const float*)d_in[8], (const float*)d_in[11]};
  const float* bb[4] = {(const float*)d_in[3], (const float*)d_in[6],
                        (const float*)d_in[9], (const float*)d_in[12]};
  const float* Wp = (const float*)d_in[13];
  const float* bp = (const float*)d_in[14];
  float* outF = (float*)d_out;

  const int ktX = LSTM_E / 32;   // 250
  const int ktH = LSTM_U / 32;   // 32
  const int ktP = LSTM_U / 32;   // 32
  const int ntG = GATES_N / 16;  // 256
  const int ntP = LSTM_V / 16;   // 500

  char* ws = (char*)d_ws;
  size_t off = 0;
  auto carve = [&](size_t bytes) { char* p = ws + off; off += (bytes + 255) & ~(size_t)255; return p; };

  __bf16* packWx = (__bf16*)carve((size_t)ntG * ktX * 512 * 2);   // 65.5 MB
  __bf16* packWh = (__bf16*)carve((size_t)ntG * ktH * 512 * 2);   //  8.4 MB
  __bf16* packWp = (__bf16*)carve((size_t)ntP * ktP * 512 * 2);   // 16.4 MB
  __bf16* outB   = (__bf16*)carve((size_t)LSTM_B * LSTM_E * 2);   //  1.0 MB
  __bf16* hB     = (__bf16*)carve((size_t)LSTM_B * LSTM_U * 2);   //  128 KB
  float*  cState = (float*)carve((size_t)LSTM_B * LSTM_U * 4);    //  256 KB
  float*  gates  = (float*)carve((size_t)LSTM_B * GATES_N * 4);   //  1.0 MB
  float*  logits = (float*)carve((size_t)LSTM_B * LSTM_V * 4);    //  2.0 MB
  float*  bias4  = (float*)carve((size_t)GATES_N * 4);            //   16 KB

  // ---- per-launch weight packing fp32 -> bf16 WMMA tiles (L2-resident) ----
  {
    long long exElems = (long long)LSTM_E * LSTM_U;   // 8,192,000
    long long ehElems = (long long)LSTM_U * LSTM_U;   // 1,048,576
    long long epElems = (long long)LSTM_U * LSTM_V;   // 8,192,000
    int bx = 256;
    for (int g = 0; g < 4; ++g) {
      pack_weights<<<(int)((exElems + bx - 1) / bx), bx, 0, stream>>>(
          Wx[g], packWx, LSTM_E, LSTM_U, g * (LSTM_U / 16), ktX, exElems);
      pack_weights<<<(int)((ehElems + bx - 1) / bx), bx, 0, stream>>>(
          Wh[g], packWh, LSTM_U, LSTM_U, g * (LSTM_U / 16), ktH, ehElems);
    }
    pack_weights<<<(int)((epElems + bx - 1) / bx), bx, 0, stream>>>(
        Wp, packWp, LSTM_U, LSTM_V, 0, ktP, epElems);
    concat_bias<<<(GATES_N + 255) / 256, 256, 0, stream>>>(bb[0], bb[1], bb[2], bb[3], bias4);
    init_state<<<(LSTM_B * LSTM_E + 255) / 256, 256, 0, stream>>>(y, outB, cState, hB);
  }

  // ---- the recurrence: 64 sequential steps ----
  const int gridGates = (ntG + 7) / 8;   // 32 blocks
  const int gridProj  = (ntP + 7) / 8;   // 63 blocks
  for (int t = 0; t < LSTM_T; ++t) {
    // gates = sigmoid(out @ Wx4 + h @ Wh4 + b4)
    wmma_gemm_bias_act<<<gridGates, 256, 0, stream>>>(
        outB, packWx, ktX, hB, packWh, ktH, bias4, gates, GATES_N, ntG, 1);
    // c = f*c + i*g ; h = tanh(c)*o
    cell_update<<<(LSTM_B * LSTM_U + 255) / 256, 256, 0, stream>>>(gates, cState, hB);
    // logits = h @ Wp + bp
    wmma_gemm_bias_act<<<gridProj, 256, 0, stream>>>(
        hB, packWp, ktP, nullptr, nullptr, 0, bp, logits, LSTM_V, ntP, 0);
    // out = softmax(logits); write d_out[:, t, :] and bf16 feedback
    softmax_row<<<LSTM_B, 256, 0, stream>>>(logits, outF, outB, t);
  }
}
